// AggSample_12171937317170
// MI455X (gfx1250) — compile-verified, matrix operand order checked by reading, not run on hardware
//
#include <hip/hip_runtime.h>
#include <hip/hip_bf16.h>
#include <math.h>

typedef __attribute__((ext_vector_type(16))) _Float16 v16h;
typedef __attribute__((ext_vector_type(8)))  _Float16 v8h;
typedef __attribute__((ext_vector_type(8)))  float    v8f;
typedef __attribute__((ext_vector_type(4)))  int      v4i;

#define NB 8
#define NC 8192
#define TH 28
#define TW 28
#define TP 784      // 28*28
#define SH 12
#define SW 12
#define SQ 144      // 12*12
#define GG 25
#define GP 625      // 25*25
#define CSPLIT 8
#define CCHUNK (NC / CSPLIT)
#define KT (NC / 32)      // 256 K-tiles of 32
#define MT 7              // M-tiles (of 16 rows) per block; 784 = 7 * 112
#define INV_TTEMP 25.0f   // 1/0.04
#define INV_STEMP 10.0f   // 1/0.1

// workspace layout (in floats)
#define OFF_M    0
#define OFF_ZINV (OFF_M    + NB * TP)
#define OFF_PM   (OFF_ZINV + NB * TP)
#define OFF_PZ   (OFF_PM   + NB * CSPLIT * TP)
#define OFF_G    (OFF_PZ   + NB * CSPLIT * TP)          // NB*TP*SQ floats (~3.6 MB)
#define OFF_SP   (OFF_G    + NB * TP * SQ)              // packed student f16: NB*NC*SQ halves (~19 MB)

// ---------------------------------------------------------------------------
// CDNA5 async global->LDS copy (ASYNCcnt path). The builtin takes int4*
// pointers (per the round-2 diagnostic); use AS-qualified casts so it works
// whether the params are AS-qualified or generic. Host pass sees no-ops.
#if defined(__HIP_DEVICE_COMPILE__)
  #if __has_builtin(__builtin_amdgcn_global_load_async_to_lds_b128)
    #define ASYNC_COPY16(gsrc, ldst)                                          \
      __builtin_amdgcn_global_load_async_to_lds_b128(                         \
          (__attribute__((address_space(1))) v4i*)(gsrc),                     \
          (__attribute__((address_space(3))) v4i*)(ldst), 0, 0)
  #else
    #define ASYNC_COPY16(gsrc, ldst)                                          \
      asm volatile("global_load_async_to_lds_b128 %0, %1, off"                \
                   :: "v"((unsigned)(size_t)(ldst)), "v"(gsrc) : "memory")
  #endif
  #if __has_builtin(__builtin_amdgcn_s_wait_asynccnt)
    #define WAIT_ASYNC() __builtin_amdgcn_s_wait_asynccnt(0)
  #else
    #define WAIT_ASYNC() asm volatile("s_wait_asynccnt 0x0" ::: "memory")
  #endif
#else
  #define ASYNC_COPY16(gsrc, ldst) ((void)0)
  #define WAIT_ASYNC()             ((void)0)
#endif

// ---------------------------------------------------------------------------
__global__ void k_zero(float* out) {
    if (blockIdx.x == 0 && threadIdx.x == 0) out[0] = 0.0f;
}

// ---------------------------------------------------------------------------
// Pack student once: f16, x10 scale, tiled [b][kt][q][kk] so kernel 2's
// B-stage is a contiguous 9216B tile (async-copy friendly).
__global__ __launch_bounds__(256)
void k_pack_student(const float* __restrict__ s, _Float16* __restrict__ Sp) {
    size_t tid = (size_t)blockIdx.x * blockDim.x + threadIdx.x;  // b*NC*SQ + c*SQ + q
    if (tid >= (size_t)NB * NC * SQ) return;
    int q = (int)(tid % SQ);
    size_t bc = tid / SQ;
    int c = (int)(bc % NC);
    int b = (int)(bc / NC);
    int kt = c >> 5, kk = c & 31;
    Sp[(((size_t)b * KT + kt) * SQ + q) * 32 + kk] = (_Float16)(s[tid] * INV_STEMP);
}

// ---------------------------------------------------------------------------
// Kernel 1a: partial online logsumexp stats of (teacher - center)/0.04 over
// channel chunks. One thread per (b, chunk, p); coalesced over p.
__global__ __launch_bounds__(256)
void k_teacher_lse_part(const float* __restrict__ t, const float* __restrict__ center,
                        float* __restrict__ pm, float* __restrict__ pz) {
    int tid = blockIdx.x * blockDim.x + threadIdx.x;   // (b*CSPLIT+s)*TP + p
    if (tid >= NB * CSPLIT * TP) return;
    int p  = tid % TP;
    int bs = tid / TP;
    int s  = bs % CSPLIT;
    int b  = bs / CSPLIT;
    const float* base = t + (size_t)b * NC * TP + (size_t)s * CCHUNK * TP + p;
    const float* cen  = center + s * CCHUNK;
    float m = -3.4e38f, z = 0.0f;
    for (int c = 0; c < CCHUNK; ++c) {
        float v  = (base[(size_t)c * TP] - cen[c]) * INV_TTEMP;
        float mn = fmaxf(m, v);
        z = z * __expf(m - mn) + __expf(v - mn);
        m = mn;
    }
    pm[tid] = m;
    pz[tid] = z;
}

// Kernel 1b: combine chunk partials -> per-(b,p) max m and 1/Z.
__global__ __launch_bounds__(256)
void k_teacher_lse_comb(const float* __restrict__ pm, const float* __restrict__ pz,
                        float* __restrict__ m_out, float* __restrict__ zinv_out) {
    int tid = blockIdx.x * blockDim.x + threadIdx.x;   // b*TP + p
    if (tid >= NB * TP) return;
    int p = tid % TP, b = tid / TP;
    float M = -3.4e38f;
    #pragma unroll
    for (int s = 0; s < CSPLIT; ++s)
        M = fmaxf(M, pm[(size_t)(b * CSPLIT + s) * TP + p]);
    float Z = 0.0f;
    #pragma unroll
    for (int s = 0; s < CSPLIT; ++s) {
        size_t idx = (size_t)(b * CSPLIT + s) * TP + p;
        Z += pz[idx] * __expf(pm[idx] - M);
    }
    m_out[tid]    = M;
    zinv_out[tid] = 1.0f / Z;
}

// ---------------------------------------------------------------------------
// Kernel 2: WMMA GEMM  G[b] = softmax-numerator(teacher)^T (784x8192) x (10*student)(8192x144),
// scaled by 1/Z_p per row. Block = (b, 7 M-tiles = 112 p-rows); 9 waves = 9 N-tiles;
// 7 WMMAs per wave per 32-K step; B tile arrives via async global->LDS copy.
__global__ __launch_bounds__(288)
void k_gemm_wmma(const float* __restrict__ t, const _Float16* __restrict__ Sp,
                 const float* __restrict__ center,
                 const float* __restrict__ mP, const float* __restrict__ zinvP,
                 float* __restrict__ G) {
    __shared__ __align__(16) _Float16 At[MT * 16][32];   // [p][k] exp-numerator, 7168 B
    __shared__ __align__(16) _Float16 St[SQ][32];        // [q][k] packed student, 9216 B
    __shared__ float mloc[MT * 16], zloc[MT * 16];

    const int b      = blockIdx.y;
    const int p0     = blockIdx.x * (MT * 16);
    const int tid    = threadIdx.x;
    const int lane   = tid & 31;
    const int w      = tid >> 5;              // wave id 0..8 == N-tile
    const int lane16 = lane & 15;
    const int sel    = (lane < 16) ? 0 : 1;   // K-half group per CDNA5 16-bit layout

    for (int e = tid; e < MT * 16; e += 288) {
        mloc[e] = mP[b * TP + p0 + e];
        zloc[e] = zinvP[b * TP + p0 + e];
    }
    __syncthreads();

    const float*    tb  = t  + (size_t)b * NC * TP;
    const _Float16* spb = Sp + (size_t)b * KT * SQ * 32;
    v8f acc[MT];
    #pragma unroll
    for (int mt = 0; mt < MT; ++mt) acc[mt] = (v8f){};

    for (int kt = 0; kt < KT; ++kt) {
        const int c0 = kt * 32;
        // ---- B tile: async global->LDS, 9216 B = 576 x 16 B (2 per thread)
        const _Float16* src = spb + (size_t)kt * SQ * 32;
        _Float16* dst = &St[0][0];
        ASYNC_COPY16(src + (size_t)tid * 8,         dst + (size_t)tid * 8);
        ASYNC_COPY16(src + (size_t)(288 + tid) * 8, dst + (size_t)(288 + tid) * 8);

        // ---- A tile: 112(p) x 32(k) exp((t - center)/T - m_p), coalesced over p
        for (int e = tid; e < MT * 512; e += 288) {
            int pp = e & 15, kk = (e >> 4) & 31, mt = e >> 9;
            int prow = mt * 16 + pp;
            const float* ta = &tb[(size_t)(c0 + kk) * TP + p0 + prow];
            if (c0 + 32 < NC) __builtin_prefetch(ta + (size_t)32 * TP, 0, 1);
            float a = __expf((*ta - center[c0 + kk]) * INV_TTEMP - mloc[prow]);
            At[prow][kk] = (_Float16)a;
        }
        WAIT_ASYNC();          // each wave waits for its own async copies
        __syncthreads();       // publish LDS across the workgroup

        // fragments per CDNA5 16-bit layout: lanes 0-15 hold K {0..7,16..23},
        // lanes 16-31 hold K {8..15,24..31}; two 16B LDS reads per fragment.
        const v8h* br = (const v8h*)(&St[w * 16 + lane16][0]);
        v8h b0 = br[sel], b1 = br[sel + 2];
        v16h bf = __builtin_shufflevector(b0, b1, 0,1,2,3,4,5,6,7,8,9,10,11,12,13,14,15);
        #pragma unroll
        for (int mt = 0; mt < MT; ++mt) {
            const v8h* ar = (const v8h*)(&At[mt * 16 + lane16][0]);
            v8h a0 = ar[sel], a1 = ar[sel + 2];
            v16h af = __builtin_shufflevector(a0, a1, 0,1,2,3,4,5,6,7,8,9,10,11,12,13,14,15);
            acc[mt] = __builtin_amdgcn_wmma_f32_16x16x32_f16(false, af, false, bf,
                                                             (short)0, acc[mt], false, false);
        }
        __syncthreads();
    }

    // C/D layout: VGPR r holds row M = r + (lane<16 ? 0 : 8), col N = lane%16
    const int rbase = (lane < 16) ? 0 : 8;
    #pragma unroll
    for (int mt = 0; mt < MT; ++mt) {
        float* Gb = G + ((size_t)b * TP + p0 + mt * 16) * SQ + w * 16 + lane16;
        #pragma unroll
        for (int r = 0; r < 8; ++r) {
            int M = rbase + r;
            Gb[(size_t)M * SQ] = acc[mt][r] * zloc[mt * 16 + M];
        }
    }
}

// ---------------------------------------------------------------------------
// Kernel 3: per grid point: logsumexp of blended student logits over channels,
// minus cross term assembled from 16 entries of G; reduce to scalar loss.
__global__ __launch_bounds__(128)
void k_loss(const float* __restrict__ s, const float* __restrict__ G,
            float* __restrict__ out) {
    int tid = blockIdx.x * blockDim.x + threadIdx.x;
    float contrib = 0.0f;
    if (tid < NB * GP) {
        int b  = tid / GP;
        int ij = tid % GP;
        int i = ij / GG, j = ij % GG;   // grid[...,0] varies with row i -> ix; [...,1] with j -> iy

        // ---- student sampling coords (box -0.8..0.8, 12x12 map)
        float gx = -0.8f + 1.6f * i / 24.0f;
        float gy = -0.8f + 1.6f * j / 24.0f;
        float fx = fminf(fmaxf(((gx + 1.0f) * SW - 1.0f) * 0.5f, 0.0f), (float)(SW - 1));
        float fy = fminf(fmaxf(((gy + 1.0f) * SH - 1.0f) * 0.5f, 0.0f), (float)(SH - 1));
        int sx0 = (int)floorf(fx), sy0 = (int)floorf(fy);
        int sx1 = min(sx0 + 1, SW - 1), sy1 = min(sy0 + 1, SH - 1);
        sx0 = max(0, min(sx0, SW - 1)); sy0 = max(0, min(sy0, SH - 1));
        float wx = fx - (float)sx0, wy = fy - (float)sy0;
        int q00 = sy0 * SW + sx0, q01 = sy0 * SW + sx1;
        int q10 = sy1 * SW + sx0, q11 = sy1 * SW + sx1;
        float w00 = (1.0f - wy) * (1.0f - wx), w01 = (1.0f - wy) * wx;
        float w10 = wy * (1.0f - wx),          w11 = wy * wx;

        // ---- online logsumexp of s_common over 8192 channels
        const float* sbase = s + (size_t)b * NC * SQ;
        float m = -3.4e38f, z = 0.0f;
        for (int c = 0; c < NC; ++c) {
            const float* pc = sbase + (size_t)c * SQ;
            float v = (w00 * pc[q00] + w01 * pc[q01] + w10 * pc[q10] + w11 * pc[q11]) * INV_STEMP;
            float mn = fmaxf(m, v);
            z = z * __expf(m - mn) + __expf(v - mn);
            m = mn;
        }
        float lse = m + __logf(z);

        // ---- teacher sampling coords (box -0.9..0.9, 28x28 map)
        float tgx = -0.9f + 1.8f * i / 24.0f;
        float tgy = -0.9f + 1.8f * j / 24.0f;
        float tfx = fminf(fmaxf(((tgx + 1.0f) * TW - 1.0f) * 0.5f, 0.0f), (float)(TW - 1));
        float tfy = fminf(fmaxf(((tgy + 1.0f) * TH - 1.0f) * 0.5f, 0.0f), (float)(TH - 1));
        int tx0 = (int)floorf(tfx), ty0 = (int)floorf(tfy);
        int tx1 = min(tx0 + 1, TW - 1), ty1 = min(ty0 + 1, TH - 1);
        tx0 = max(0, min(tx0, TW - 1)); ty0 = max(0, min(ty0, TH - 1));
        float twx = tfx - (float)tx0, twy = tfy - (float)ty0;

        int   txi[2] = { tx0, tx1 };        float wtx[2] = { 1.0f - twx, twx };
        int   tyi[2] = { ty0, ty1 };        float wty[2] = { 1.0f - twy, twy };
        int   sxi[2] = { sx0, sx1 };        float wsx[2] = { 1.0f - wx, wx };
        int   syi[2] = { sy0, sy1 };        float wsy[2] = { 1.0f - wy, wy };

        // ---- cross term: 4 teacher corners x 4 student corners from G
        const float* Gb = G + (size_t)b * TP * SQ;
        float cross = 0.0f;
        #pragma unroll
        for (int dy = 0; dy < 2; ++dy)
        #pragma unroll
        for (int dx = 0; dx < 2; ++dx) {
            const float* Gp = Gb + (size_t)(tyi[dy] * TW + txi[dx]) * SQ;
            float wt = wty[dy] * wtx[dx];
            #pragma unroll
            for (int ey = 0; ey < 2; ++ey)
            #pragma unroll
            for (int ex = 0; ex < 2; ++ex)
                cross += wt * wsy[ey] * wsx[ex] * Gp[syi[ey] * SW + sxi[ex]];
        }
        contrib = lse - cross;   // (uses sum_c t_common == 1)
    }

    __shared__ float red[128];
    red[threadIdx.x] = contrib;
    __syncthreads();
    for (int st = 64; st > 0; st >>= 1) {
        if (threadIdx.x < st) red[threadIdx.x] += red[threadIdx.x + st];
        __syncthreads();
    }
    if (threadIdx.x == 0) atomicAdd(out, red[0] * (1.0f / (float)(NB * GP)));
}

// ---------------------------------------------------------------------------
extern "C" void kernel_launch(void* const* d_in, const int* in_sizes, int n_in,
                              void* d_out, int out_size, void* d_ws, size_t ws_size,
                              hipStream_t stream) {
    const float* teacher = (const float*)d_in[0];   // (8, 8192, 28, 28)
    const float* student = (const float*)d_in[1];   // (8, 8192, 12, 12)
    const float* center  = (const float*)d_in[2];   // (1, 8192)
    float* out = (float*)d_out;
    float* ws  = (float*)d_ws;

    float*     m    = ws + OFF_M;
    float*     zinv = ws + OFF_ZINV;
    float*     pm   = ws + OFF_PM;
    float*     pz   = ws + OFF_PZ;
    float*     G    = ws + OFF_G;
    _Float16*  Sp   = (_Float16*)(ws + OFF_SP);

    k_zero<<<1, 1, 0, stream>>>(out);
    {
        size_t n = (size_t)NB * NC * SQ;
        k_pack_student<<<(unsigned)((n + 255) / 256), 256, 0, stream>>>(student, Sp);
    }
    k_teacher_lse_part<<<(NB * CSPLIT * TP + 255) / 256, 256, 0, stream>>>(teacher, center, pm, pz);
    k_teacher_lse_comb<<<(NB * TP + 255) / 256, 256, 0, stream>>>(pm, pz, m, zinv);
    k_gemm_wmma<<<dim3(TP / (MT * 16), NB), 288, 0, stream>>>(teacher, Sp, center, m, zinv, G);
    k_loss<<<(NB * GP + 127) / 128, 128, 0, stream>>>(student, G, out);
}
